// PredictiveCodingLayer_24739011625049
// MI455X (gfx1250) — compile-verified, matrix operand order checked by reading, not run on hardware
//
#include <hip/hip_runtime.h>
#include <cstdint>
#include <cstddef>

typedef __attribute__((ext_vector_type(16))) __bf16 v16bf;
typedef __attribute__((ext_vector_type(8)))  __bf16 v8bf;
typedef __attribute__((ext_vector_type(8)))  float  v8f;

#define DEV_INLINE __device__ __forceinline__

constexpr int D_DIM = 1024;
constexpr int F_DIM = 256;
constexpr int S_DIM = 4096;

// ---------------- helpers ----------------
DEV_INLINE float sigmoidf_(float x){ return 1.f / (1.f + expf(-x)); }
DEV_INLINE float clamp1(float x){ return fminf(1.f, fmaxf(-1.f, x)); }

// CDNA5 async copy: 16B global -> LDS, tracked by ASYNCcnt (no VGPR round-trip).
// LDS byte address = low 32 bits of the generic pointer.
DEV_INLINE void async_copy_b128(const void* gptr, void* lptr){
  unsigned lds = (unsigned)(uintptr_t)lptr;
  unsigned long long ga = (unsigned long long)(uintptr_t)gptr;
  asm volatile("global_load_async_to_lds_b128 %0, %1, off"
               :: "v"(lds), "v"(ga) : "memory");
}
DEV_INLINE void wait_async0(){
  asm volatile("s_wait_asynccnt 0" ::: "memory");
}

DEV_INLINE float block_sum_256(float v, float* sred, int tid){
  #pragma unroll
  for (int off = 16; off > 0; off >>= 1) v += __shfl_down(v, off, 32);
  if ((tid & 31) == 0) sred[tid >> 5] = v;
  __syncthreads();
  if (tid == 0){
    float t = 0.f;
    #pragma unroll
    for (int i = 0; i < 8; ++i) t += sred[i];
    sred[0] = t;
  }
  __syncthreads();
  return sred[0];
}

// ---------------- small prep kernels ----------------
__global__ void f2bf_kernel(const float* __restrict__ src, __bf16* __restrict__ dst, int n){
  int i = blockIdx.x * blockDim.x + threadIdx.x;
  if (i < n) dst[i] = (__bf16)src[i];
}

__global__ void coeff_kernel(const float* __restrict__ logd, const float* __restrict__ freq,
                             float* __restrict__ arot){
  int f = threadIdx.x;
  if (f < F_DIM){
    float decay = 1.f / (1.f + expf(-logd[f]));
    float omega = tanhf(freq[f]) * 0.1f;
    float c = cosf(omega), s = sinf(omega);
    arot[f]           = decay * c;   // A.re
    arot[F_DIM + f]   = decay * s;   // A.im
    arot[2*F_DIM + f] = c;           // rot.re
    arot[3*F_DIM + f] = s;           // rot.im
  }
}

// ---------------- layernorm (one block per row of D=1024) ----------------
__global__ __launch_bounds__(256) void layernorm_kernel(
    const float* __restrict__ x, const float* __restrict__ g, const float* __restrict__ bta,
    float* __restrict__ outF, __bf16* __restrict__ outB)
{
  __shared__ float sred[8];
  const int tid = threadIdx.x;
  const float* xr = x + (size_t)blockIdx.x * D_DIM;
  float v[4];
  float s = 0.f;
  #pragma unroll
  for (int i = 0; i < 4; ++i){ v[i] = xr[tid + i*256]; s += v[i]; }
  s = block_sum_256(s, sred, tid);
  const float mu = s * (1.f / D_DIM);
  float q = 0.f;
  #pragma unroll
  for (int i = 0; i < 4; ++i){ float d = v[i] - mu; q += d * d; }
  __syncthreads();
  q = block_sum_256(q, sred, tid);
  const float rstd = rsqrtf(q * (1.f / D_DIM) + 1e-5f);
  #pragma unroll
  for (int i = 0; i < 4; ++i){
    int c = tid + i*256;
    float y = (v[i] - mu) * rstd * g[c] + bta[c];
    if (outF) outF[(size_t)blockIdx.x * D_DIM + c] = y;
    if (outB) outB[(size_t)blockIdx.x * D_DIM + c] = (__bf16)y;
  }
}

// ---------------- complex linear-recurrence scan (replaces FFT conv) ----------------
// y[t] = rot * h[t],  h[t] = A*h[t-1] + u[t];  one thread per (b, f)
__global__ __launch_bounds__(256) void scan_kernel(
    const float* __restrict__ spec, const float* __restrict__ arot,
    __bf16* __restrict__ sout, int S)
{
  const int f = threadIdx.x;     // 0..255
  const int b = blockIdx.x;
  const float Are = arot[f],           Aim = arot[F_DIM + f];
  const float Rre = arot[2*F_DIM + f], Rim = arot[3*F_DIM + f];
  const int F2 = 2 * F_DIM;
  unsigned idx = (unsigned)b * (unsigned)S * F2 + f;
  float hre = 0.f, him = 0.f;
  float ure = spec[idx], uim = spec[idx + F_DIM];   // preload t=0
  for (int t = 0; t < S; ++t){
    float nre = 0.f, nim = 0.f;
    if (t + 1 < S){ nre = spec[idx + F2]; nim = spec[idx + F2 + F_DIM]; }  // prefetch
    float tre = fmaf(Are, hre, fmaf(-Aim, him, ure));
    float tim = fmaf(Are, him, fmaf( Aim, hre, uim));
    hre = tre; him = tim;
    sout[idx]         = (__bf16)(Rre*hre - Rim*him);
    sout[idx + F_DIM] = (__bf16)(Rre*him + Rim*hre);
    idx += F2;
    ure = nre; uim = nim;
  }
}

// ---------------- bf16 WMMA GEMM:  C[M,N] = A[M,K] * W[N,K]^T ----------------
// Double-buffered: async global->LDS copies for slab i+1 overlap WMMA on slab i.
enum { EP_NONE = 0, EP_ERR = 1, EP_GELU = 2, EP_BIAS = 3 };

template<int EPI>
__global__ __launch_bounds__(256) void gemm_bf16_wmma(
    const __bf16* __restrict__ A, const __bf16* __restrict__ W,
    int M, int N, int K,
    float* __restrict__ outF, __bf16* __restrict__ outB,
    const float* __restrict__ bias,
    const float* __restrict__ td, const float* __restrict__ ew,
    float* __restrict__ partial)
{
  constexpr int BM = 64, BN = 128, BK = 32, LR = BK + 8; // 80B row stride: 16B aligned, conflict-free
  __shared__ __bf16 As[2][BM][LR];
  __shared__ __bf16 Bs[2][BN][LR];
  __shared__ float  red[8];

  const int tid  = threadIdx.x;
  const int lane = tid & 31;
  const int wave = tid >> 5;     // 8 waves
  const int wm = wave >> 2;      // 0..1 -> 32 rows each
  const int wn = wave & 3;       // 0..3 -> 32 cols each
  const unsigned bm0 = blockIdx.x * BM;
  const unsigned bn0 = blockIdx.y * BN;

  // global->LDS staging (256 threads move 64x32 + 128x32 bf16 per K-slab via async DMA)
  const int arow = tid >> 2, acol = (tid & 3) * 8;    // one 16B packet
  const int brow = tid >> 1, bcol = (tid & 1) * 16;   // two 16B packets
  const __bf16* gA = A + (size_t)(bm0 + arow) * K + acol;
  const __bf16* gB = W + (size_t)(bn0 + brow) * K + bcol;

  // WMMA 16-bit operand lane mapping (ISA 7.12.2): lane<16 holds K {0..7,16..23}, lane>=16 holds K {8..15,24..31}
  const int fr = lane & 15;
  const int kb = (lane < 16) ? 0 : 8;

  v8f acc[2][2] = {};

  const int nslab = K / BK;
  auto issue = [&](int buf, int ks){
    const int ko = ks * BK;
    async_copy_b128(gA + ko,     &As[buf][arow][acol]);
    async_copy_b128(gB + ko,     &Bs[buf][brow][bcol]);
    async_copy_b128(gB + ko + 8, &Bs[buf][brow][bcol + 8]);
  };

  issue(0, 0);   // prime the pipeline
  for (int i = 0; i < nslab; ++i){
    wait_async0();        // slab i's copies (only ones outstanding) have landed
    __syncthreads();      // ... for every thread in the block
    if (i + 1 < nslab) issue((i + 1) & 1, i + 1);  // overlap next transfer with compute

    const int buf = i & 1;
    v16bf af[2], bfr[2];
    #pragma unroll
    for (int tm = 0; tm < 2; ++tm){
      const int r = wm*32 + tm*16 + fr;
      v8bf lo = *(const v8bf*)(&As[buf][r][kb]);
      v8bf hi = *(const v8bf*)(&As[buf][r][kb + 16]);
      af[tm] = __builtin_shufflevector(lo, hi, 0,1,2,3,4,5,6,7,8,9,10,11,12,13,14,15);
    }
    #pragma unroll
    for (int tn = 0; tn < 2; ++tn){
      const int r = wn*32 + tn*16 + fr;
      v8bf lo = *(const v8bf*)(&Bs[buf][r][kb]);
      v8bf hi = *(const v8bf*)(&Bs[buf][r][kb + 16]);
      bfr[tn] = __builtin_shufflevector(lo, hi, 0,1,2,3,4,5,6,7,8,9,10,11,12,13,14,15);
    }
    #pragma unroll
    for (int tm = 0; tm < 2; ++tm)
      #pragma unroll
      for (int tn = 0; tn < 2; ++tn)
        acc[tm][tn] = __builtin_amdgcn_wmma_f32_16x16x32_bf16(
            false, af[tm], false, bfr[tn], (short)0, acc[tm][tn], false, false);
  }

  // ---- epilogue (32-bit indexing: all flat indices < 2^31) ----
  float esum = 0.f;
  const float sw = (EPI == EP_ERR) ? sigmoidf_(ew[0]) : 0.f;
  const int mofs = (lane < 16) ? 0 : 8;   // C/D layout: lanes>=16 hold M+8
  #pragma unroll
  for (int tm = 0; tm < 2; ++tm){
    #pragma unroll
    for (int tn = 0; tn < 2; ++tn){
      const unsigned gn = bn0 + wn*32 + tn*16 + fr;
      const unsigned gmBase = bm0 + wm*32 + tm*16 + mofs;
      #pragma unroll
      for (int v = 0; v < 8; ++v){
        const unsigned idx = (gmBase + v) * (unsigned)N + gn;
        float c = acc[tm][tn][v];
        if constexpr (EPI == EP_NONE){
          outF[idx] = c;
        } else if constexpr (EPI == EP_ERR){
          float e = clamp1(c - td[idx]);
          float corr = c - sw * e;
          outF[idx] = corr;
          outB[idx] = (__bf16)corr;
          esum += e * e;
        } else if constexpr (EPI == EP_GELU){
          float xv = c + bias[gn];
          float gl = 0.5f * xv * (1.f + erff(xv * 0.70710678118f));
          outB[idx] = (__bf16)gl;
        } else { // EP_BIAS
          outF[idx] = c + bias[gn];
        }
      }
    }
  }
  if constexpr (EPI == EP_ERR){
    #pragma unroll
    for (int off = 16; off > 0; off >>= 1) esum += __shfl_down(esum, off, 32);
    if (lane == 0) red[wave] = esum;
    __syncthreads();
    if (tid == 0){
      float t = 0.f;
      #pragma unroll
      for (int i = 0; i < 8; ++i) t += red[i];
      partial[blockIdx.x * gridDim.y + blockIdx.y] = t;  // one deterministic slot per block
    }
  }
}

// ---------------- deterministic fixed-order error reduction ----------------
__global__ void finalize_err(const float* __restrict__ partial, int n, float inv_cnt,
                             float* __restrict__ out){
  if (threadIdx.x == 0 && blockIdx.x == 0){
    float s = 0.f;
    for (int i = 0; i < n; ++i) s += partial[i];
    s *= inv_cnt;
    out[0] = fminf(1.f, fmaxf(0.f, s));
  }
}

// ---------------- host orchestration ----------------
extern "C" void kernel_launch(void* const* d_in, const int* in_sizes, int n_in,
                              void* d_out, int out_size, void* d_ws, size_t ws_size,
                              hipStream_t stream)
{
  (void)n_in; (void)out_size; (void)ws_size;
  const float* x_bu  = (const float*)d_in[0];
  const float* x_td  = (const float*)d_in[1];
  const float* Wspec = (const float*)d_in[2];
  const float* logd  = (const float*)d_in[3];
  const float* freq  = (const float*)d_in[4];
  const float* Wfrom = (const float*)d_in[5];
  const float* ln1g  = (const float*)d_in[6];
  const float* ln1b  = (const float*)d_in[7];
  const float* W1    = (const float*)d_in[8];
  const float* b1    = (const float*)d_in[9];
  const float* W2    = (const float*)d_in[10];
  const float* b2    = (const float*)d_in[11];
  const float* ln2g  = (const float*)d_in[12];
  const float* ln2b  = (const float*)d_in[13];
  const float* ew    = (const float*)d_in[14];

  const int M  = in_sizes[0] / D_DIM;   // B*S = 16384
  const int Bb = M / S_DIM;             // 4
  const int F2 = 2 * F_DIM;             // 512

  char* ws = (char*)d_ws;
  size_t o = 0;
  auto take = [&](size_t bytes) -> void* {
    void* p = ws + o; o += (bytes + 255) & ~(size_t)255; return p;
  };

  __bf16* xn_bf    = (__bf16*)take((size_t)M * D_DIM * 2);  // later reused as h1_bf
  float*  spec_f   = (float*) take((size_t)M * F2 * 4);     // later reused as corr_bf (same size)
  __bf16* sout_bf  = (__bf16*)take((size_t)M * F2 * 2);
  float*  h2_f     = (float*) take((size_t)M * D_DIM * 4);
  __bf16* wspec_bf = (__bf16*)take((size_t)F2 * D_DIM * 2);
  __bf16* wfrom_bf = (__bf16*)take((size_t)D_DIM * F2 * 2);
  __bf16* w1_bf    = (__bf16*)take((size_t)D_DIM * D_DIM * 2);
  __bf16* w2_bf    = (__bf16*)take((size_t)D_DIM * D_DIM * 2);
  float*  arot     = (float*) take((size_t)4 * F_DIM * 4);
  const int nblkM  = M / 64;
  float*  partial  = (float*) take((size_t)nblkM * (D_DIM / 128) * 4);

  __bf16* corr_bf = (__bf16*)spec_f;  // spec consumed by scan before this is written
  __bf16* h1_bf   = xn_bf;            // xn consumed by GEMM1 before this is written

  float* out_corr = (float*)d_out;
  float* out_pred = out_corr + (size_t)M * D_DIM;
  float* out_err  = out_pred + (size_t)M * D_DIM;

  // 0) weight precision conversion + recurrence coefficients
  f2bf_kernel<<<(F2*D_DIM + 255)/256, 256, 0, stream>>>(Wspec, wspec_bf, F2*D_DIM);
  f2bf_kernel<<<(D_DIM*F2 + 255)/256, 256, 0, stream>>>(Wfrom, wfrom_bf, D_DIM*F2);
  f2bf_kernel<<<(D_DIM*D_DIM + 255)/256, 256, 0, stream>>>(W1, w1_bf, D_DIM*D_DIM);
  f2bf_kernel<<<(D_DIM*D_DIM + 255)/256, 256, 0, stream>>>(W2, w2_bf, D_DIM*D_DIM);
  coeff_kernel<<<1, 256, 0, stream>>>(logd, freq, arot);

  // 1) layernorm + cast
  layernorm_kernel<<<M, 256, 0, stream>>>(x_bu, ln1g, ln1b, nullptr, xn_bf);

  // 2) spectral = xn @ W_spec^T   [M,1024]x[512,1024]^T
  gemm_bf16_wmma<EP_NONE><<<dim3(nblkM, F2/128), 256, 0, stream>>>(
      xn_bf, wspec_bf, M, F2, D_DIM, spec_f, nullptr, nullptr, nullptr, nullptr, nullptr);

  // 3) causal conv == complex linear recurrence
  scan_kernel<<<Bb, F_DIM, 0, stream>>>(spec_f, arot, sout_bf, S_DIM);

  // 4) bu_repr = sout @ W_from^T, fused error/corrected epilogue
  gemm_bf16_wmma<EP_ERR><<<dim3(nblkM, D_DIM/128), 256, 0, stream>>>(
      sout_bf, wfrom_bf, M, D_DIM, F2, out_corr, corr_bf, nullptr, x_td, ew, partial);

  // 5) h1 = gelu(corrected @ W1^T + b1)
  gemm_bf16_wmma<EP_GELU><<<dim3(nblkM, D_DIM/128), 256, 0, stream>>>(
      corr_bf, w1_bf, M, D_DIM, D_DIM, nullptr, h1_bf, b1, nullptr, nullptr, nullptr);

  // 6) h2 = h1 @ W2^T + b2
  gemm_bf16_wmma<EP_BIAS><<<dim3(nblkM, D_DIM/128), 256, 0, stream>>>(
      h1_bf, w2_bf, M, D_DIM, D_DIM, h2_f, nullptr, b2, nullptr, nullptr, nullptr);

  // 7) next_prediction = layernorm(h2)
  layernorm_kernel<<<M, 256, 0, stream>>>(h2_f, ln2g, ln2b, out_pred, nullptr);

  // 8) prediction_error (deterministic fixed-order sum)
  finalize_err<<<1, 32, 0, stream>>>(partial, nblkM * (D_DIM/128),
                                     1.f / ((float)M * (float)D_DIM), out_err);
}
